// Dist_loss_60773787238931
// MI455X (gfx1250) — matrix operand order repaired and checked by reading, not compiled, not run on hardware
//
#include <hip/hip_runtime.h>
#include <hip/hip_fp16.h>
#include <math.h>

// ---------------------------------------------------------------------------
// Problem constants (reference: x1,x2 = (8192,256) f32, w = (256,) f32)
// ---------------------------------------------------------------------------
#define N_PTS 8192
#define DIMS  256
#define LOG2E 1.44269504088896340736f
#define LN2   0.69314718055994530942f

typedef _Float16 half8  __attribute__((ext_vector_type(8)));
typedef _Float16 half16 __attribute__((ext_vector_type(16)));
typedef float    f32x4  __attribute__((ext_vector_type(4)));
typedef float    f32x8  __attribute__((ext_vector_type(8)));

// Native base-2 transcendentals (v_exp_f32 / v_log_f32 are exp2/log2 in HW).
__device__ __forceinline__ float fexp2(float x) { return __builtin_amdgcn_exp2f(x); }
__device__ __forceinline__ float flog2(float x) { return __builtin_amdgcn_logf(x); }

// ---------------------------------------------------------------------------
// 1) w_c = clip(w,0,2); w_norm = w_c / mean(w_c)     (one 256-thread block)
// ---------------------------------------------------------------------------
__global__ void prep_w_kernel(const float* __restrict__ w,
                              float* __restrict__ w_norm) {
    __shared__ float red[256];
    const int t = threadIdx.x;
    float wc = fminf(fmaxf(w[t], 0.0f), 2.0f);
    red[t] = wc;
    __syncthreads();
    for (int s = 128; s > 0; s >>= 1) {
        if (t < s) red[t] += red[t + s];
        __syncthreads();
    }
    const float mean = red[0] * (1.0f / 256.0f);
    w_norm[t] = wc / mean;
}

// ---------------------------------------------------------------------------
// 2) Row-normalize a = w_norm*x1 and b = x2 into f16 (one block per row).
// ---------------------------------------------------------------------------
__global__ void normalize_kernel(const float* __restrict__ x1,
                                 const float* __restrict__ x2,
                                 const float* __restrict__ w_norm,
                                 _Float16* __restrict__ an,
                                 _Float16* __restrict__ bn) {
    __shared__ float red[256];
    const int t = threadIdx.x;
    const int r = blockIdx.x;
    float v;
    _Float16* dst;
    if (r < N_PTS) {
        v   = w_norm[t] * x1[(size_t)r * DIMS + t];
        dst = an + (size_t)r * DIMS;
    } else {
        v   = x2[(size_t)(r - N_PTS) * DIMS + t];
        dst = bn + (size_t)(r - N_PTS) * DIMS;
    }
    red[t] = v * v;
    __syncthreads();
    for (int s = 128; s > 0; s >>= 1) {
        if (t < s) red[t] += red[t + s];
        __syncthreads();
    }
    const float rn = 1.0f / (sqrtf(red[0]) + 1e-12f);
    dst[t] = (_Float16)(v * rn);
}

// ---------------------------------------------------------------------------
// 3) Cost GEMM: C[i][j] = 1 - dot(A_row_i, B_row_j) via v_wmma_f32_16x16x32_f16
//    Wave tile 16(M)x64(N); 8 waves stack in M -> 128x64 block tile; K
//    unrolled 8x32 -> 32 WMMAs.  A/B k order: contiguous k = 16*(lane/16) + s.
// ---------------------------------------------------------------------------
__global__ __launch_bounds__(256, 2)
void gemm_cost_kernel(const _Float16* __restrict__ A,
                      const _Float16* __restrict__ B,
                      float* __restrict__ C) {
    const int lane  = threadIdx.x & 31;
    const int wave  = threadIdx.x >> 5;
    const int sel   = lane >> 4;
    const int l16   = lane & 15;
    const int mtile = blockIdx.y * 128 + wave * 16;
    const int ntile = blockIdx.x * 64;

    f32x8 acc0 = {}, acc1 = {}, acc2 = {}, acc3 = {};

    const _Float16* __restrict__ arow = A + (size_t)(mtile + l16) * DIMS;
    const _Float16* __restrict__ b0r  = B + (size_t)(ntile +  0 + l16) * DIMS;
    const _Float16* __restrict__ b1r  = B + (size_t)(ntile + 16 + l16) * DIMS;
    const _Float16* __restrict__ b2r  = B + (size_t)(ntile + 32 + l16) * DIMS;
    const _Float16* __restrict__ b3r  = B + (size_t)(ntile + 48 + l16) * DIMS;

#pragma unroll
    for (int koff = 0; koff < DIMS; koff += 32) {
        half8 alo = *(const half8*)(arow + koff + sel * 16);
        half8 ahi = *(const half8*)(arow + koff + sel * 16 + 8);
        half16 afrag = __builtin_shufflevector(alo, ahi,
                          0,1,2,3,4,5,6,7,8,9,10,11,12,13,14,15);

        const int boff = koff + sel * 16;
        half16 bf0 = *(const half16*)(b0r + boff);
        half16 bf1 = *(const half16*)(b1r + boff);
        half16 bf2 = *(const half16*)(b2r + boff);
        half16 bf3 = *(const half16*)(b3r + boff);

        acc0 = __builtin_amdgcn_wmma_f32_16x16x32_f16(false, afrag, false, bf0,
                                                      (short)0, acc0, false, false);
        acc1 = __builtin_amdgcn_wmma_f32_16x16x32_f16(false, afrag, false, bf1,
                                                      (short)0, acc1, false, false);
        acc2 = __builtin_amdgcn_wmma_f32_16x16x32_f16(false, afrag, false, bf2,
                                                      (short)0, acc2, false, false);
        acc3 = __builtin_amdgcn_wmma_f32_16x16x32_f16(false, afrag, false, bf3,
                                                      (short)0, acc3, false, false);
    }

    const int m0 = mtile + sel * 8;
#pragma unroll
    for (int r = 0; r < 8; ++r) {
        float* crow = C + (size_t)(m0 + r) * N_PTS + l16;
        crow[ntile +  0] = 1.0f - acc0[r];
        crow[ntile + 16] = 1.0f - acc1[r];
        crow[ntile + 32] = 1.0f - acc2[r];
        crow[ntile + 48] = 1.0f - acc3[r];
    }
}

// ---------------------------------------------------------------------------
// Wave32 cross-lane (max,sum) LSE combine via ds-permute shuffles.
// ---------------------------------------------------------------------------
__device__ __forceinline__ void wave_lse_reduce(float& m, float& s) {
#pragma unroll
    for (int off = 16; off > 0; off >>= 1) {
        const float m2 = __shfl_xor(m, off, 32);
        const float s2 = __shfl_xor(s, off, 32);
        const float mn = fmaxf(m, m2);
        s = s * fexp2(m - mn) + s2 * fexp2(m2 - mn);
        m = mn;
    }
}

// ---------------------------------------------------------------------------
// 4) Plain row softmin (used for symmetric matrices Cxx, Cyy):
//    out[i] = -eps * LSE_j( logw + (pot[j] - C[i][j])/eps )
//    Base-2 online LSE, micro-batched 4-wide (1.5 exp2/element).
// ---------------------------------------------------------------------------
__global__ __launch_bounds__(256)
void softmin_kernel(const float* __restrict__ C,
                    const float* __restrict__ pot,
                    float* __restrict__ out,
                    float eps, float sc /* log2e/eps */, float logw) {
    __shared__ float sm[8];
    __shared__ float ss[8];
    const int t    = threadIdx.x;
    const int lane = t & 31;
    const int wv   = t >> 5;
    const float* __restrict__ crow = C + (size_t)blockIdx.x * N_PTS;

    float m = -3.4e38f, s = 0.0f;
#pragma unroll
    for (int i = 0; i < 8; ++i) {
        const int j4 = (i * 256 + t) * 4;
        f32x4 c = *(const f32x4*)(crow + j4);
        f32x4 p = *(const f32x4*)(pot + j4);
        float u0 = (p[0] - c[0]) * sc, u1 = (p[1] - c[1]) * sc;
        float u2 = (p[2] - c[2]) * sc, u3 = (p[3] - c[3]) * sc;
        const float tm = fmaxf(fmaxf(u0, u1), fmaxf(u2, u3));
        const float p4 = fexp2(u0 - tm) + fexp2(u1 - tm) +
                         fexp2(u2 - tm) + fexp2(u3 - tm);
        const float mn = fmaxf(m, tm);
        s = s * fexp2(m - mn) + p4 * fexp2(tm - mn);
        m = mn;
    }
    wave_lse_reduce(m, s);
    if (lane == 0) { sm[wv] = m; ss[wv] = s; }
    __syncthreads();
    if (t == 0) {
        float M = sm[0], S = ss[0];
#pragma unroll
        for (int w = 1; w < 8; ++w) {
            const float mn = fmaxf(M, sm[w]);
            S = S * fexp2(M - mn) + ss[w] * fexp2(sm[w] - mn);
            M = mn;
        }
        out[blockIdx.x] = -eps * (logw + LN2 * (M + flog2(S)));
    }
}

// ---------------------------------------------------------------------------
// 5) Fused row+column softmin over Cxy (single 256MB pass per iteration):
//      row side:  f_new[i] = -eps*LSE_j(logw + (g[j] - C[i][j])/eps)
//      col side:  per-block partial (max,sum) of (f[i] - C[i][j])/eps per col
//    Block = 256 threads handles a 32-row stripe; thread owns 32 fixed
//    columns -> g-potentials cached/pre-scaled in registers, column
//    accumulators live in registers, spilled once as per-stripe partials.
// ---------------------------------------------------------------------------
__global__ __launch_bounds__(256)
void softmin_rowcol_kernel(const float* __restrict__ C,
                           const float* __restrict__ potc,  // g  (column pot)
                           const float* __restrict__ potr,  // f  (row pot)
                           float* __restrict__ out_rows,    // f_new
                           float* __restrict__ part_m,      // (gridDim.x, 8192)
                           float* __restrict__ part_s,
                           float eps, float sc /* log2e/eps */, float logw) {
    __shared__ float sm[8];
    __shared__ float ss[8];
    const int t    = threadIdx.x;
    const int lane = t & 31;
    const int wv   = t >> 5;
    const int rb   = blockIdx.x;           // 32-row stripe index

    // Pre-scaled column potentials for this thread's 32 columns.
    float pg[32];
#pragma unroll
    for (int i = 0; i < 8; ++i) {
        f32x4 p = *(const f32x4*)(potc + (i * 256 + t) * 4);
#pragma unroll
        for (int k = 0; k < 4; ++k) pg[i * 4 + k] = p[k] * sc;
    }
    // Column LSE accumulators (registers).
    float cm[32], cs[32];
#pragma unroll
    for (int k = 0; k < 32; ++k) { cm[k] = -3.4e38f; cs[k] = 0.0f; }

#pragma unroll 1
    for (int r = 0; r < 32; ++r) {
        const size_t row = (size_t)rb * 32 + r;
        const float* __restrict__ crow = C + row * N_PTS;
        if (r < 31)  // hint next row into cache while we churn this one
            __builtin_prefetch(crow + N_PTS + t * 4, 0, 0);
        const float fr = potr[row] * sc;

        float m = -3.4e38f, s = 0.0f;
#pragma unroll
        for (int i = 0; i < 8; ++i) {
            f32x4 c = *(const f32x4*)(crow + (i * 256 + t) * 4);
            float u[4];
#pragma unroll
            for (int k = 0; k < 4; ++k) {
                const float cc  = c[k] * sc;
                u[k] = pg[i * 4 + k] - cc;          // row-side argument
                const float v   = fr - cc;          // col-side argument
                const int   idx = i * 4 + k;
                const float mn  = fmaxf(cm[idx], v);
                cs[idx] = cs[idx] * fexp2(cm[idx] - mn) + fexp2(v - mn);
                cm[idx] = mn;
            }
            const float tm = fmaxf(fmaxf(u[0], u[1]), fmaxf(u[2], u[3]));
            const float p4 = fexp2(u[0] - tm) + fexp2(u[1] - tm) +
                             fexp2(u[2] - tm) + fexp2(u[3] - tm);
            const float mn = fmaxf(m, tm);
            s = s * fexp2(m - mn) + p4 * fexp2(tm - mn);
            m = mn;
        }
        wave_lse_reduce(m, s);
        if (lane == 0) { sm[wv] = m; ss[wv] = s; }
        __syncthreads();
        if (t == 0) {
            float M = sm[0], S = ss[0];
#pragma unroll
            for (int w = 1; w < 8; ++w) {
                const float mn = fmaxf(M, sm[w]);
                S = S * fexp2(M - mn) + ss[w] * fexp2(sm[w] - mn);
                M = mn;
            }
            out_rows[row] = -eps * (logw + LN2 * (M + flog2(S)));
        }
        __syncthreads();
    }

    // Spill per-stripe column partials.
    float* pm = part_m + (size_t)rb * N_PTS;
    float* ps = part_s + (size_t)rb * N_PTS;
#pragma unroll
    for (int i = 0; i < 8; ++i) {
        const int j4 = (i * 256 + t) * 4;
        f32x4 vm, vs;
#pragma unroll
        for (int k = 0; k < 4; ++k) { vm[k] = cm[i * 4 + k]; vs[k] = cs[i * 4 + k]; }
        *(f32x4*)(pm + j4) = vm;
        *(f32x4*)(ps + j4) = vs;
    }
}

// ---------------------------------------------------------------------------
// 6) Combine the 256 per-stripe column partials -> g_new[j]
// ---------------------------------------------------------------------------
__global__ __launch_bounds__(256)
void col_combine_kernel(const float* __restrict__ part_m,
                        const float* __restrict__ part_s,
                        float* __restrict__ out_cols,
                        float eps, float logw) {
    const int j = blockIdx.x * 256 + threadIdx.x;
    float m = -3.4e38f, s = 0.0f;
#pragma unroll 4
    for (int b = 0; b < N_PTS / 32; ++b) {
        const float m2 = part_m[(size_t)b * N_PTS + j];
        const float s2 = part_s[(size_t)b * N_PTS + j];
        const float mn = fmaxf(m, m2);
        s = s * fexp2(m - mn) + s2 * fexp2(m2 - mn);
        m = mn;
    }
    out_cols[j] = -eps * (logw + LN2 * (m + flog2(s)));
}

// ---------------------------------------------------------------------------
// 7) Small helpers
// ---------------------------------------------------------------------------
__global__ void avg_kernel(float* __restrict__ pot,
                           const float* __restrict__ upd) {
    const int i = blockIdx.x * blockDim.x + threadIdx.x;
    pot[i] = 0.5f * (pot[i] + upd[i]);
}

__global__ void zero4_kernel(float* a, float* b, float* c, float* d) {
    const int i = blockIdx.x * blockDim.x + threadIdx.x;
    a[i] = 0.0f; b[i] = 0.0f; c[i] = 0.0f; d[i] = 0.0f;
}

__global__ void final_reduce_kernel(const float* __restrict__ ff,
                                    const float* __restrict__ fxxf,
                                    const float* __restrict__ gf,
                                    const float* __restrict__ gyyf,
                                    float* __restrict__ out) {
    __shared__ float red[256];
    const int t = threadIdx.x;
    float s = 0.0f;
    for (int i = t; i < N_PTS; i += 256)
        s += (ff[i] - fxxf[i]) + (gf[i] - gyyf[i]);
    red[t] = s;
    __syncthreads();
    for (int st = 128; st > 0; st >>= 1) {
        if (t < st) red[t] += red[t + st];
        __syncthreads();
    }
    if (t == 0) out[0] = red[0] * (1.0f / (float)N_PTS);
}

// ---------------------------------------------------------------------------
// Host orchestration (graph-capture safe: everything on `stream`)
// ---------------------------------------------------------------------------
extern "C" void kernel_launch(void* const* d_in, const int* in_sizes, int n_in,
                              void* d_out, int out_size, void* d_ws, size_t ws_size,
                              hipStream_t stream) {
    (void)in_sizes; (void)n_in; (void)out_size; (void)ws_size;
    const float* x1 = (const float*)d_in[0];
    const float* x2 = (const float*)d_in[1];
    const float* w  = (const float*)d_in[2];
    float* out = (float*)d_out;

    char*  ws  = (char*)d_ws;
    size_t off = 0;
    auto alloc = [&](size_t bytes) -> void* {
        void* p = ws + off;
        off += (bytes + 255) & ~(size_t)255;
        return p;
    };
    const size_t cbytes = (size_t)N_PTS * N_PTS * sizeof(float);        // 256 MB
    const size_t pbytes = (size_t)(N_PTS / 32) * N_PTS * sizeof(float); // 8 MB
    float*    Cxy = (float*)alloc(cbytes);
    float*    Cxx = (float*)alloc(cbytes);
    float*    Cyy = (float*)alloc(cbytes);
    float* part_m = (float*)alloc(pbytes);
    float* part_s = (float*)alloc(pbytes);
    _Float16* an  = (_Float16*)alloc((size_t)N_PTS * DIMS * sizeof(_Float16));
    _Float16* bn  = (_Float16*)alloc((size_t)N_PTS * DIMS * sizeof(_Float16));
    float* w_norm = (float*)alloc(DIMS * sizeof(float));
    const size_t pb = N_PTS * sizeof(float);
    float* f      = (float*)alloc(pb);
    float* g      = (float*)alloc(pb);
    float* fxx    = (float*)alloc(pb);
    float* gyy    = (float*)alloc(pb);
    float* tf     = (float*)alloc(pb);
    float* tg     = (float*)alloc(pb);
    float* ffin   = (float*)alloc(pb);
    float* gfin   = (float*)alloc(pb);
    float* fxxfin = (float*)alloc(pb);
    float* gyyfin = (float*)alloc(pb);

    // --- Preprocess + cost matrices -------------------------------------
    prep_w_kernel<<<1, 256, 0, stream>>>(w, w_norm);
    normalize_kernel<<<2 * N_PTS, 256, 0, stream>>>(x1, x2, w_norm, an, bn);

    const dim3 ggrid(N_PTS / 64, N_PTS / 128);
    gemm_cost_kernel<<<ggrid, 256, 0, stream>>>(an, bn, Cxy);
    gemm_cost_kernel<<<ggrid, 256, 0, stream>>>(an, an, Cxx);
    gemm_cost_kernel<<<ggrid, 256, 0, stream>>>(bn, bn, Cyy);

    zero4_kernel<<<N_PTS / 256, 256, 0, stream>>>(f, g, fxx, gyy);

    // --- Sinkhorn eps schedule ------------------------------------------
    const float logw = -logf((float)N_PTS);
    float eps_list[16];
    int   ne = 0;
    for (double e = 4.0; e > 0.0025; e *= 0.25) eps_list[ne++] = (float)e;
    for (int i = 0; i < 4; ++i)                eps_list[ne++] = 0.0025f;

    const int rcgrid = N_PTS / 32;   // stripes for fused kernel
    for (int it = 0; it < ne; ++it) {
        const float eps = eps_list[it], sc = LOG2E / eps;
        // debiasing potentials: averaged fixed-point update (row passes only,
        // Cxx and Cyy are symmetric)
        softmin_kernel<<<N_PTS, 256, 0, stream>>>(Cxx, fxx, tf, eps, sc, logw);
        avg_kernel<<<N_PTS / 256, 256, 0, stream>>>(fxx, tf);
        softmin_kernel<<<N_PTS, 256, 0, stream>>>(Cyy, gyy, tg, eps, sc, logw);
        avg_kernel<<<N_PTS / 256, 256, 0, stream>>>(gyy, tg);
        // cross potentials: Jacobi update, ONE pass over Cxy for both f and g
        softmin_rowcol_kernel<<<rcgrid, 256, 0, stream>>>(
            Cxy, g, f, tf, part_m, part_s, eps, sc, logw);
        col_combine_kernel<<<N_PTS / 256, 256, 0, stream>>>(
            part_m, part_s, tg, eps, logw);
        float* tmp;
        tmp = f; f = tf; tf = tmp;
        tmp = g; g = tg; tg = tmp;
    }

    // --- Final extrapolation + divergence -------------------------------
    const float eps = 0.0025f, sc = LOG2E / eps;
    softmin_rowcol_kernel<<<rcgrid, 256, 0, stream>>>(
        Cxy, g, f, ffin, part_m, part_s, eps, sc, logw);
    col_combine_kernel<<<N_PTS / 256, 256, 0, stream>>>(
        part_m, part_s, gfin, eps, logw);
    softmin_kernel<<<N_PTS, 256, 0, stream>>>(Cxx, fxx, fxxfin, eps, sc, logw);
    softmin_kernel<<<N_PTS, 256, 0, stream>>>(Cyy, gyy, gyyfin, eps, sc, logw);
    final_reduce_kernel<<<1, 256, 0, stream>>>(ffin, fxxfin, gfin, gyyfin, out);
}
